// RandomFeaturesAttentionHead_24704651886870
// MI455X (gfx1250) — matrix operand order repaired and checked by reading, not compiled, not run on hardware
//
#include <hip/hip_runtime.h>
#include <hip/hip_bf16.h>

#define BB 16
#define TT 4096
#define HH 768
#define DD 64
#define MM 256

typedef __bf16 v16bf __attribute__((ext_vector_type(16)));
typedef float  v8f   __attribute__((ext_vector_type(8)));
typedef unsigned int u32x4 __attribute__((ext_vector_type(4)));
typedef int          i32x4 __attribute__((ext_vector_type(4)));
typedef int          i32x8 __attribute__((ext_vector_type(8)));

union BFrag {
    v16bf v;
    unsigned short u[16];
    uint4 q[2];
};

static __device__ __forceinline__ unsigned short f2bf(float f) {
    unsigned int u = __float_as_uint(f);
    u = u + 0x7FFFu + ((u >> 16) & 1u);   // round-to-nearest-even
    return (unsigned short)(u >> 16);
}
static __device__ __forceinline__ float bf2f(unsigned short h) {
    return __uint_as_float(((unsigned int)h) << 16);
}
static __device__ __forceinline__ v8f v8zero() {
    v8f z;
#pragma unroll
    for (int i = 0; i < 8; ++i) z[i] = 0.0f;
    return z;
}
static __device__ __forceinline__ v8f wmma_bf16(v16bf a, v16bf b, v8f c) {
    return __builtin_amdgcn_wmma_f32_16x16x32_bf16(false, a, false, b, (short)0, c, false, false);
}
static __device__ __forceinline__ void pack4(unsigned short* u, float4 f) {
    u[0] = f2bf(f.x); u[1] = f2bf(f.y); u[2] = f2bf(f.z); u[3] = f2bf(f.w);
}

// ---------------------------------------------------------------------------
// Tensor Data Mover: 1D (dword) tile copy global -> LDS.
// D# per CDNA5 ISA ch.8: group0 = {count=1 | lds_addr | global_addr | type=2},
// group1 = {data_size=4B, tensor_dim0, tensor_dim1=1, tile_dim0, tile_dim1=1,
//           tensor_dim0_stride}.
// ---------------------------------------------------------------------------
static __device__ __forceinline__ void tdm_load_1d(unsigned lds_off, const void* gptr,
                                                   unsigned tile_dwords) {
    unsigned long long ga = (unsigned long long)(uintptr_t)gptr;
    u32x4 g0;
    g0[0] = 1u;                                              // count=1, user desc
    g0[1] = lds_off;                                         // lds_addr (bytes)
    g0[2] = (unsigned)(ga & 0xFFFFFFFFu);                    // global_addr[31:0]
    g0[3] = (unsigned)((ga >> 32) & 0x01FFFFFFu) | (2u << 30);  // addr[56:32] | type=2
    i32x8 g1;
    g1[0] = (int)(2u << 16);                                 // data_size=2 (4-byte)
    g1[1] = (int)((tile_dwords & 0xFFFFu) << 16);            // tensor_dim0[15:0]
    g1[2] = (int)(((tile_dwords >> 16) & 0xFFFFu) | (1u << 16)); // td0[31:16], tensor_dim1=1
    g1[3] = (int)((tile_dwords & 0xFFFFu) << 16);            // td1[31:16]=0, tile_dim0
    g1[4] = (int)1u;                                         // tile_dim1=1, tile_dim2=0
    g1[5] = (int)tile_dwords;                                // tensor_dim0_stride[31:0]
    g1[6] = 0;
    g1[7] = 0;
    i32x4 z4 = {0, 0, 0, 0};
#if defined(__clang_major__) && (__clang_major__ >= 23)
    i32x8 z8 = {0, 0, 0, 0, 0, 0, 0, 0};
    __builtin_amdgcn_tensor_load_to_lds(g0, g1, z4, z4, z8, 0);
#else
    __builtin_amdgcn_tensor_load_to_lds(g0, g1, z4, z4, 0);
#endif
}

// ---------------------------------------------------------------------------
// k0: repack weights to bf16 fragment-friendly layouts, zero accumulators.
//   Wq/Wk/Wv [H,D] -> wqkv16[hc][mat][d][32]  (hc = h>>5, K-in-chunk contiguous)
//   w  [M,D]       -> wf16[(d>>5)*256 + m][32]
// ---------------------------------------------------------------------------
__global__ void k0_prep(const float* __restrict__ Wq, const float* __restrict__ Wk,
                        const float* __restrict__ Wv, const float* __restrict__ w,
                        unsigned short* wqkv16, unsigned short* wf16,
                        float* ksum, float* kptv) {
    const int NW = HH * DD;  // 49152
    int i = blockIdx.x * blockDim.x + threadIdx.x;
    if (i < 3 * NW) {
        const int mat = (i < NW) ? 0 : (i < 2 * NW ? 1 : 2);
        const float* src = (mat == 0) ? Wq : (mat == 1 ? Wk : Wv);
        int j = i - mat * NW;
        int h = j / DD, d = j - h * DD;
        // chunk-major: [(h>>5)*3 + mat][d][h&31]
        wqkv16[(((h >> 5) * 3 + mat) * DD + d) * 32 + (h & 31)] = f2bf(src[j]);
        return;
    }
    i -= 3 * NW;
    if (i < MM * DD) {
        int m = i / DD, d = i - m * DD;
        wf16[((d >> 5) * MM + m) * 32 + (d & 31)] = f2bf(w[i]);
        return;
    }
    i -= MM * DD;
    if (i < BB * MM) { ksum[i] = 0.0f; return; }
    i -= BB * MM;
    if (i < BB * DD * MM) { kptv[i] = 0.0f; }
}

// ---------------------------------------------------------------------------
// k1: QKV projection. One wave = 16 tokens; K-loop over H in chunks of 32.
// Weight chunks (12KB: 3 matrices x 64 cols x 32 K) are TDM-double-buffered
// into LDS by wave 0; all waves consume B fragments from LDS.
// Outputs q,k fp32 [B,T,D] and v bf16 transposed [B,D,T].
// ---------------------------------------------------------------------------
__global__ void __launch_bounds__(256) k1_qkv(
    const float* __restrict__ x,
    const float* __restrict__ bq, const float* __restrict__ bk, const float* __restrict__ bv,
    const unsigned short* __restrict__ wqkv16,
    float* __restrict__ qout, float* __restrict__ kout, unsigned short* __restrict__ vT) {
    __shared__ __align__(64) unsigned short wbuf[2][3 * DD * 32];  // 2 x 12KB
    const int lane = threadIdx.x & 31;
    const int wave = threadIdx.x >> 5;
    const int half = lane >> 4;
    const int n    = lane & 15;
    const int b    = blockIdx.x >> 5;      // T/128 = 32 chunks per batch
    const int tch  = blockIdx.x & 31;
    const int t0   = tch * 128 + wave * 16;
    const int NCH  = HH / 32;              // 24
    const unsigned CH_DW = 3 * DD * 32 * 2 / 4;  // 3072 dwords per chunk

    v8f accQ[4], accK[4], accV[4];
#pragma unroll
    for (int dt = 0; dt < 4; ++dt) { accQ[dt] = v8zero(); accK[dt] = v8zero(); accV[dt] = v8zero(); }

    const float* xrow = x + ((size_t)b * TT + t0 + n) * HH;  // A row = token t0+n

    if (wave == 0)
        tdm_load_1d((unsigned)(uintptr_t)(&wbuf[0][0]), wqkv16, CH_DW);

    for (int hc = 0; hc < NCH; ++hc) {
        if (wave == 0) {
            if (hc + 1 < NCH) {
                tdm_load_1d((unsigned)(uintptr_t)(&wbuf[(hc + 1) & 1][0]),
                            wqkv16 + (size_t)(hc + 1) * 3 * DD * 32, CH_DW);
                __builtin_amdgcn_s_wait_tensorcnt(1);  // chunk hc has landed (in-order)
            } else {
                __builtin_amdgcn_s_wait_tensorcnt(0);
            }
        }
        __syncthreads();

        const int h0 = hc * 32;
        if (h0 + 64 < HH) __builtin_prefetch(xrow + h0 + 64, 0, 0);
        // A fragment: elems 0..15 -> K = half*16+e
        BFrag A;
        const float* g0 = xrow + h0 + half * 16;
        const float* g1 = xrow + h0 + half * 16 + 8;
        pack4(&A.u[0],  *(const float4*)(g0));
        pack4(&A.u[4],  *(const float4*)(g0 + 4));
        pack4(&A.u[8],  *(const float4*)(g1));
        pack4(&A.u[12], *(const float4*)(g1 + 4));

        const unsigned short* wb = &wbuf[hc & 1][0];
#pragma unroll
        for (int dt = 0; dt < 4; ++dt) {
            const int col = dt * 16 + n;
            v16bf Bq = *(const v16bf*)(wb + (0 * DD + col) * 32 + half * 16);
            v16bf Bk = *(const v16bf*)(wb + (1 * DD + col) * 32 + half * 16);
            v16bf Bv = *(const v16bf*)(wb + (2 * DD + col) * 32 + half * 16);
            accQ[dt] = wmma_bf16(A.v, Bq, accQ[dt]);
            accK[dt] = wmma_bf16(A.v, Bk, accK[dt]);
            accV[dt] = wmma_bf16(A.v, Bv, accV[dt]);
        }
        __syncthreads();  // all waves done reading before buffer is overwritten
    }
    // Epilogue: C layout lane(half,n), VGPR v -> (token = t0+half*8+v, d = dt*16+n)
#pragma unroll
    for (int dt = 0; dt < 4; ++dt) {
        const int col = dt * 16 + n;
        const float biq = bq[col], bik = bk[col], biv = bv[col];
        union { uint4 q; unsigned short u[8]; } vpk;
#pragma unroll
        for (int v = 0; v < 8; ++v) {
            const int t = t0 + half * 8 + v;
            const size_t o = ((size_t)b * TT + t) * DD + col;
            qout[o] = accQ[dt][v] + biq;
            kout[o] = accK[dt][v] + bik;
            vpk.u[v] = f2bf(accV[dt][v] + biv);
        }
        *(uint4*)(vT + ((size_t)b * DD + col) * TT + t0 + half * 8) = vpk.q;
    }
}

// ---------------------------------------------------------------------------
// k2: qp = exp(q w^T - 0.5||q||^2)/sqrt(M), same for kp. Wave = 16 tokens.
// Random-feature weights (32KB) TDM-loaded to LDS once. qp stored [B,T,M],
// kp stored transposed [B,M,T]; kp_sum accumulated with fp32 atomics.
// ---------------------------------------------------------------------------
__global__ void __launch_bounds__(256) k2_feat(
    const float* __restrict__ qf, const float* __restrict__ kf,
    const unsigned short* __restrict__ wf16,
    unsigned short* __restrict__ qp, unsigned short* __restrict__ kpT,
    float* __restrict__ ksum) {
    __shared__ __align__(64) unsigned short swf[2 * MM * 32];  // 32KB
    __shared__ float sxd[8 * 16];
    const int lane = threadIdx.x & 31;
    const int wave = threadIdx.x >> 5;
    const int half = lane >> 4;
    const int n    = lane & 15;
    const int b    = blockIdx.x >> 5;
    const int tch  = blockIdx.x & 31;
    const int t0   = tch * 128 + wave * 16;
    const float invSqrtM = 0.0625f;  // 1/sqrt(256)

    if (wave == 0) {
        tdm_load_1d((unsigned)(uintptr_t)(&swf[0]), wf16, 2 * MM * 32 * 2 / 4);
        __builtin_amdgcn_s_wait_tensorcnt(0);
    }
    __syncthreads();

    for (int pass = 0; pass < 2; ++pass) {
        const float* zrow = (pass ? kf : qf) + ((size_t)b * TT + t0 + n) * DD;
        BFrag A0, A1;
        float xd = 0.0f;
#pragma unroll
        for (int kc = 0; kc < 2; ++kc) {
            const float* p0 = zrow + kc * 32 + half * 16;
            const float* p1 = zrow + kc * 32 + half * 16 + 8;
            float4 f0 = *(const float4*)(p0);
            float4 f1 = *(const float4*)(p0 + 4);
            float4 f2 = *(const float4*)(p1);
            float4 f3 = *(const float4*)(p1 + 4);
            BFrag& A = kc ? A1 : A0;
            pack4(&A.u[0], f0); pack4(&A.u[4], f1); pack4(&A.u[8], f2); pack4(&A.u[12], f3);
            xd += f0.x*f0.x + f0.y*f0.y + f0.z*f0.z + f0.w*f0.w
                + f1.x*f1.x + f1.y*f1.y + f1.z*f1.z + f1.w*f1.w
                + f2.x*f2.x + f2.y*f2.y + f2.z*f2.z + f2.w*f2.w
                + f3.x*f3.x + f3.y*f3.y + f3.z*f3.z + f3.w*f3.w;
        }
        xd *= 0.5f;
        xd += __shfl_xor(xd, 16);
        __syncthreads();
        if (half == 0) sxd[wave * 16 + n] = xd;
        __syncthreads();
        float xdv[8];
#pragma unroll
        for (int v = 0; v < 8; ++v) xdv[v] = sxd[wave * 16 + half * 8 + v];

        for (int mt = 0; mt < 16; ++mt) {
            const int m = mt * 16 + n;
            v16bf B0 = *(const v16bf*)(swf + (0 * MM + m) * 32 + half * 16);
            v16bf B1 = *(const v16bf*)(swf + (1 * MM + m) * 32 + half * 16);
            v8f acc = v8zero();
            acc = wmma_bf16(A0.v, B0, acc);
            acc = wmma_bf16(A1.v, B1, acc);
            if (pass == 0) {
#pragma unroll
                for (int v = 0; v < 8; ++v) {
                    const int t = t0 + half * 8 + v;
                    const float val = __expf(acc[v] - xdv[v]) * invSqrtM;
                    qp[((size_t)b * TT + t) * MM + m] = f2bf(val);
                }
            } else {
                union { uint4 q; unsigned short u[8]; } pk;
                float s = 0.0f;
#pragma unroll
                for (int v = 0; v < 8; ++v) {
                    const float val = __expf(acc[v] - xdv[v]) * invSqrtM;
                    pk.u[v] = f2bf(val);
                    s += val;
                }
                *(uint4*)(kpT + ((size_t)b * MM + m) * TT + t0 + half * 8) = pk.q;
                s += __shfl_xor(s, 16);
                if (half == 0) atomicAdd(&ksum[b * MM + m], s);
            }
        }
    }
}

// ---------------------------------------------------------------------------
// k3: kptv[b,d,m] = sum_t v[b,t,d]*kp[b,t,m]. WMMA K-dim = tokens; both
// operands stored transposed so all fragment loads are contiguous.
// ---------------------------------------------------------------------------
__global__ void __launch_bounds__(256) k3_kptv(
    const unsigned short* __restrict__ vT, const unsigned short* __restrict__ kpT,
    float* __restrict__ kptv) {
    const int lane = threadIdx.x & 31;
    const int wave = threadIdx.x >> 5;
    const int half = lane >> 4;
    const int n    = lane & 15;
    const int b      = blockIdx.x >> 2;
    const int tstart = (blockIdx.x & 3) * 1024;
    const int mt0    = wave * 2;

    v8f acc[4][2];
#pragma unroll
    for (int dt = 0; dt < 4; ++dt)
#pragma unroll
        for (int j = 0; j < 2; ++j) acc[dt][j] = v8zero();

    for (int tk = tstart; tk < tstart + 1024; tk += 32) {
        BFrag A[4];
#pragma unroll
        for (int dt = 0; dt < 4; ++dt) {
            const int d = dt * 16 + n;
            const unsigned short* base = vT + ((size_t)b * DD + d) * TT + tk;
            A[dt].q[0] = *(const uint4*)(base + half * 16);
            A[dt].q[1] = *(const uint4*)(base + half * 16 + 8);
        }
        v16bf Bf[2];
#pragma unroll
        for (int j = 0; j < 2; ++j) {
            const int m = (mt0 + j) * 16 + n;
            Bf[j] = *(const v16bf*)(kpT + ((size_t)b * MM + m) * TT + tk + half * 16);
        }
#pragma unroll
        for (int dt = 0; dt < 4; ++dt)
#pragma unroll
            for (int j = 0; j < 2; ++j)
                acc[dt][j] = wmma_bf16(A[dt].v, Bf[j], acc[dt][j]);
    }
#pragma unroll
    for (int dt = 0; dt < 4; ++dt)
#pragma unroll
        for (int j = 0; j < 2; ++j)
#pragma unroll
            for (int v = 0; v < 8; ++v) {
                const int d = dt * 16 + half * 8 + v;
                const int m = (mt0 + j) * 16 + n;
                atomicAdd(&kptv[((size_t)b * DD + d) * MM + m], acc[dt][j][v]);
            }
}

// ---------------------------------------------------------------------------
// k4: Dn[t] = qp[t,:].ksum ; attn[t,d] = (qp[t,:].kptv[d,:]) / Dn[t]
// ---------------------------------------------------------------------------
__global__ void __launch_bounds__(256) k4_out(
    const unsigned short* __restrict__ qp, const float* __restrict__ kptv,
    const float* __restrict__ ksum, float* __restrict__ attn) {
    __shared__ float sdn[8 * 16];
    const int lane = threadIdx.x & 31;
    const int wave = threadIdx.x >> 5;
    const int half = lane >> 4;
    const int n    = lane & 15;
    const int b    = blockIdx.x >> 5;
    const int tch  = blockIdx.x & 31;
    const int t0   = tch * 128 + wave * 16;

    const unsigned short* qrow = qp + ((size_t)b * TT + t0 + n) * MM;
    const float* ks = ksum + b * MM;
    float dn = 0.0f;
    for (int m = half * 128; m < half * 128 + 128; m += 8) {
        union { uint4 q; unsigned short u[8]; } pk;
        pk.q = *(const uint4*)(qrow + m);
#pragma unroll
        for (int j = 0; j < 8; ++j) dn += bf2f(pk.u[j]) * ks[m + j];
    }
    dn += __shfl_xor(dn, 16);
    if (half == 0) sdn[wave * 16 + n] = dn;
    __syncthreads();
    float dnv[8];
#pragma unroll
    for (int v = 0; v < 8; ++v) dnv[v] = sdn[wave * 16 + half * 8 + v];

    v8f acc[4];
#pragma unroll
    for (int dt = 0; dt < 4; ++dt) acc[dt] = v8zero();

    for (int mc = 0; mc < 8; ++mc) {
        BFrag A;
        A.q[0] = *(const uint4*)(qrow + mc * 32 + half * 16);
        A.q[1] = *(const uint4*)(qrow + mc * 32 + half * 16 + 8);
#pragma unroll
        for (int dt = 0; dt < 4; ++dt) {
            const int d = dt * 16 + n;
            const float* kr = kptv + ((size_t)b * DD + d) * MM + mc * 32 + half * 16;
            BFrag Bf;
            pack4(&Bf.u[0],  *(const float4*)(kr));
            pack4(&Bf.u[4],  *(const float4*)(kr + 4));
            pack4(&Bf.u[8],  *(const float4*)(kr + 8));
            pack4(&Bf.u[12], *(const float4*)(kr + 12));
            acc[dt] = wmma_bf16(A.v, Bf.v, acc[dt]);
        }
    }
#pragma unroll
    for (int dt = 0; dt < 4; ++dt)
#pragma unroll
        for (int v = 0; v < 8; ++v) {
            const int t = t0 + half * 8 + v;
            attn[((size_t)b * TT + t) * DD + dt * 16 + n] = acc[dt][v] / dnv[v];
        }
}

// ---------------------------------------------------------------------------
extern "C" void kernel_launch(void* const* d_in, const int* in_sizes, int n_in,
                              void* d_out, int out_size, void* d_ws, size_t ws_size,
                              hipStream_t stream) {
    const float* x  = (const float*)d_in[0];
    const float* Wq = (const float*)d_in[1];
    const float* bq = (const float*)d_in[2];
    const float* Wk = (const float*)d_in[3];
    const float* bk = (const float*)d_in[4];
    const float* Wv = (const float*)d_in[5];
    const float* bv = (const float*)d_in[6];
    const float* w  = (const float*)d_in[7];

    float* out  = (float*)d_out;
    float* attn = out;                                 // [B,T,D]
    float* kptv = out + (size_t)BB * TT * DD;          // [B,D,M]

    char* ws = (char*)d_ws;
    size_t off = 0;
    float* qf = (float*)(ws + off);              off += (size_t)BB * TT * DD * 4;  // 16 MB
    float* kf = (float*)(ws + off);              off += (size_t)BB * TT * DD * 4;  // 16 MB
    unsigned short* vT     = (unsigned short*)(ws + off); off += (size_t)BB * DD * TT * 2;  // 8 MB
    unsigned short* qpb    = (unsigned short*)(ws + off); off += (size_t)BB * TT * MM * 2;  // 32 MB
    unsigned short* kpT    = (unsigned short*)(ws + off); off += (size_t)BB * MM * TT * 2;  // 32 MB
    unsigned short* wqkv16 = (unsigned short*)(ws + off); off += (size_t)3 * HH * DD * 2;
    unsigned short* wf16   = (unsigned short*)(ws + off); off += (size_t)MM * DD * 2;
    float* ksm = (float*)(ws + off);             off += (size_t)BB * MM * 4;
    (void)ws_size; (void)in_sizes; (void)n_in; (void)out_size;

    const int n0 = 3 * HH * DD + MM * DD + BB * MM + BB * DD * MM;
    k0_prep<<<(n0 + 255) / 256, 256, 0, stream>>>(Wq, Wk, Wv, w, wqkv16, wf16, ksm, kptv);
    k1_qkv<<<BB * (TT / 128), 256, 0, stream>>>(x, bq, bk, bv, wqkv16, qf, kf, vT);
    k2_feat<<<BB * (TT / 128), 256, 0, stream>>>(qf, kf, wf16, qpb, kpT, ksm);
    k3_kptv<<<BB * (TT / 1024), 256, 0, stream>>>(vT, kpT, kptv);
    k4_out<<<BB * (TT / 128), 256, 0, stream>>>(qpb, kptv, ksm, attn);
}
